// SOM_4569845203078
// MI455X (gfx1250) — matrix operand order repaired and checked by reading, not compiled, not run on hardware
//
#include <hip/hip_runtime.h>
#include <hip/hip_bf16.h>
#include <math.h>

// ---------------- problem constants ----------------
#define BB 4096          // batch
#define KK 4096          // codebook size
#define DD 1024          // feature dim
#define LR 0.3f
#define DSBETA 1e-4f
#define ACOEF (LR * DSBETA)          // 3e-5
#define EPS_DS 0.01f
#define NCHUNK 8                     // K split per b-tile for parallel argmax
#define KCHUNK (KK / NCHUNK)         // 512
#define BMU_WPB 2                    // waves per block in bmu kernel
#define BTILE 32                     // b columns per wave (2 x 16-wide subtiles)

typedef __attribute__((ext_vector_type(16))) __bf16 v16bf;
typedef __attribute__((ext_vector_type(8)))  __bf16 v8bf;
typedef __attribute__((ext_vector_type(8)))  float  v8f;
typedef __attribute__((ext_vector_type(4)))  unsigned int u32x4;
typedef __attribute__((ext_vector_type(8)))  int i32x8;
typedef __attribute__((ext_vector_type(4)))  int i32x4;

__device__ __forceinline__ unsigned short f2bf(float f) {
    unsigned u = __float_as_uint(f);
    unsigned r = u + 0x7FFFu + ((u >> 16) & 1u);      // round-to-nearest-even
    if ((u & 0x7F800000u) == 0x7F800000u) r = u;      // inf/nan passthrough
    return (unsigned short)(r >> 16);
}

__device__ __forceinline__ float wave_sum(float v) {
    #pragma unroll
    for (int off = 16; off > 0; off >>= 1) v += __shfl_down(v, off, 32);
    return v;
}

// A-frag (16x32 bf16, M=k, K=d): lane m<16 holds row m halves {0..7,16..23};
//                                lane m+16 holds row m halves {8..15,24..31}.
#define LOAD_A(dst, base, off) do {                                   \
    *((v8bf*)&(dst))       = *(const v8bf*)((base) + (off));          \
    *(((v8bf*)&(dst)) + 1) = *(const v8bf*)((base) + (off) + 16); } while (0)
// B-frag (32x16 bf16, K=d, N=b): lane n<16 holds col n K {0..7,16..23};
//                                lane n+16 holds col n K {8..15,24..31}.
#define LOAD_B(dst, base, off) do {                                   \
    *((v8bf*)&(dst))       = *(const v8bf*)((base) + (off));          \
    *(((v8bf*)&(dst)) + 1) = *(const v8bf*)((base) + (off) + 16); } while (0)

#define WMMA_BF16(A, Bv, C) \
    __builtin_amdgcn_wmma_f32_16x16x32_bf16(false, (A), false, (Bv), (short)0, (C), false, false)

// ---------------------------------------------------------------------------
// Kernel 1: row norms / relevance sums + bf16 conversion of x and weights.
// One wave32 per row; rows [0,KK) = weights rows, rows [KK,KK+BB) = x rows.
// ---------------------------------------------------------------------------
__global__ __launch_bounds__(128) void som_prep(
    const float* __restrict__ x, const float* __restrict__ w,
    const float* __restrict__ rel,
    unsigned short* __restrict__ xb, unsigned short* __restrict__ wb,
    float* __restrict__ x_norm, float* __restrict__ w_norm,
    float* __restrict__ rel_sum)
{
    int lane = threadIdx.x & 31;
    int wid  = (blockIdx.x * blockDim.x + threadIdx.x) >> 5;
    if (wid < KK) {
        const float* wr = w   + (size_t)wid * DD;
        const float* rr = rel + (size_t)wid * DD;
        unsigned short* dst = wb + (size_t)wid * DD;
        float s = 0.f, rs = 0.f;
        #pragma unroll 4
        for (int j = 0; j < DD / 32; ++j) {
            int d = lane + 32 * j;
            float e = wr[d];
            s  += e * e;
            rs += rr[d];
            dst[d] = f2bf(e);
        }
        s  = wave_sum(s);
        rs = wave_sum(rs);
        if (lane == 0) { w_norm[wid] = s; rel_sum[wid] = rs; }
    } else if (wid < KK + BB) {
        int b = wid - KK;
        const float* xr = x + (size_t)b * DD;
        unsigned short* dst = xb + (size_t)b * DD;
        float s = 0.f;
        #pragma unroll 4
        for (int j = 0; j < DD / 32; ++j) {
            int d = lane + 32 * j;
            float e = xr[d];
            s += e * e;
            dst[d] = f2bf(e);
        }
        s = wave_sum(s);
        if (lane == 0) x_norm[b] = s;
    }
}

// ---------------------------------------------------------------------------
// Kernel 2: WMMA distance GEMM + streaming BMU argmax.
// Output mapping: M = k (rows), N = b (cols). Each wave owns 32 b-columns
// (two 16-wide subtiles sharing every A fragment -> half the weight traffic),
// one 32x1024 bf16 x-tile in LDS loaded via the Tensor Data Mover.
// The D-loop is software-pipelined (double-buffered fragments) so WMMAs can
// overlap the global/LDS fills instead of hitting loadcnt==0 stalls.
// ---------------------------------------------------------------------------
__global__ __launch_bounds__(64) void som_bmu(
    const unsigned short* __restrict__ xb, const unsigned short* __restrict__ wb,
    const float* __restrict__ x_norm, const float* __restrict__ w_norm,
    const float* __restrict__ rel_sum,
    float* __restrict__ pv, int* __restrict__ pi)
{
    extern __shared__ unsigned short lds_all[];
    const int lane  = threadIdx.x & 31;
    const int wib   = threadIdx.x >> 5;
    const int wid   = blockIdx.x * BMU_WPB + wib;
    const int btile = wid / NCHUNK;
    const int chunk = wid % NCHUNK;
    const int b0    = btile * BTILE;
    const int kbeg  = chunk * KCHUNK;
    const int kend  = kbeg + KCHUNK;

    unsigned short* ldsx = lds_all + (size_t)wib * BTILE * DD;   // 64KB / wave

    // ---- stage 32 x-rows (64KB bf16) into LDS via the Tensor Data Mover ----
#if defined(__has_builtin) && __has_builtin(__builtin_amdgcn_tensor_load_to_lds)
    {
        // 2D Tensor DMA descriptor: data_size=2B, tile 1024 x 32, stride0=1024
        unsigned lds_off = (unsigned)(size_t)(void*)ldsx;          // LDS byte addr
        unsigned long long ga = (unsigned long long)(size_t)(xb + (size_t)b0 * DD);
        u32x4 g0 = { 1u,                                           // count=1
                     lds_off,                                      // lds_addr
                     (unsigned)(ga & 0xFFFFFFFFu),                 // global_addr lo
                     (unsigned)((ga >> 32) & 0x01FFFFFFu) | (2u << 30) }; // hi | type=2
        i32x8 g1 = { (int)0x00010000u,        // wg_mask=0, data_size=1 (2 bytes)
                     (int)(1024u << 16),      // tensor_dim0 = 1024
                     (int)(32u << 16),        // tensor_dim1 = 32
                     (int)(1024u << 16),      // tile_dim0 = 1024
                     (int)32,                 // tile_dim1 = 32, tile_dim2 = 0
                     (int)1024,               // tensor_dim0_stride = 1024
                     0, 0 };                  // tensor_dim1_stride = 0
        i32x4 z4 = { 0, 0, 0, 0 };
        i32x8 z8 = { 0, 0, 0, 0, 0, 0, 0, 0 };
        __builtin_amdgcn_tensor_load_to_lds(g0, g1, z4, z4, z8, 0);
        __builtin_amdgcn_s_wait_tensorcnt(0);
    }
#else
    {
        const uint4* src = (const uint4*)(xb + (size_t)b0 * DD);
        uint4* dst = (uint4*)ldsx;
        #pragma unroll 4
        for (int i = lane; i < BTILE * DD / 8; i += 32) dst[i] = src[i];
    }
#endif

    const int col   = lane & 15;     // output column within subtile; also A row m
    const int hi    = lane >> 4;     // 0 or 1
    const int koffA = hi * 8;        // A half offset AND C/D row offset
    const float xn0 = x_norm[b0 + col];
    const float xn1 = x_norm[b0 + 16 + col];

    const __bf16* ldsb0 = (const __bf16*)(ldsx) + (size_t)col * DD + hi * 8;
    const __bf16* ldsb1 = ldsb0 + (size_t)16 * DD;
    const __bf16* wbb   = (const __bf16*)wb;

    float best0 = -1e30f, best1 = -1e30f;
    int   bidx0 = 0,      bidx1 = 0;

    for (int kt = kbeg; kt < kend; kt += 16) {
        v8f acc0 = {0.f, 0.f, 0.f, 0.f, 0.f, 0.f, 0.f, 0.f};
        v8f acc1 = {0.f, 0.f, 0.f, 0.f, 0.f, 0.f, 0.f, 0.f};
        const __bf16* wrow = wbb + (size_t)(kt + col) * DD + koffA;

        if (kt + 16 < kend)
            __builtin_prefetch(wbb + (size_t)(kt + 16 + col) * DD + koffA, 0, 1);

        // software-pipelined D loop: double-buffered fragments
        v16bf a0, a1, p0, p1, q0, q1;
        LOAD_A(a0, wrow, 0);
        LOAD_B(p0, ldsb0, 0);
        LOAD_B(q0, ldsb1, 0);
        int d0 = 0;
        for (; d0 < DD - 64; d0 += 64) {
            LOAD_A(a1, wrow, d0 + 32);
            LOAD_B(p1, ldsb0, d0 + 32);
            LOAD_B(q1, ldsb1, d0 + 32);
            acc0 = WMMA_BF16(a0, p0, acc0);
            acc1 = WMMA_BF16(a0, q0, acc1);
            LOAD_A(a0, wrow, d0 + 64);
            LOAD_B(p0, ldsb0, d0 + 64);
            LOAD_B(q0, ldsb1, d0 + 64);
            acc0 = WMMA_BF16(a1, p1, acc0);
            acc1 = WMMA_BF16(a1, q1, acc1);
        }
        LOAD_A(a1, wrow, d0 + 32);
        LOAD_B(p1, ldsb0, d0 + 32);
        LOAD_B(q1, ldsb1, d0 + 32);
        acc0 = WMMA_BF16(a0, p0, acc0);
        acc1 = WMMA_BF16(a0, q0, acc1);
        acc0 = WMMA_BF16(a1, p1, acc0);
        acc1 = WMMA_BF16(a1, q1, acc1);

        const int kbase = kt + koffA;
        #pragma unroll
        for (int r = 0; r < 8; ++r) {
            int k = kbase + r;
            float wn = w_norm[k];
            float rs = rel_sum[k];
            // subtile 0
            float dist = xn0 + wn - 2.0f * acc0[r];
            if (!(dist == dist)) dist = 0.f;
            float dw = dist * rs * (1.0f / (float)DD);
            if (!(dw == dw)) dw = 0.f;
            float act = rs / (rs + dw + 1e-7f);
            if (act > best0) { best0 = act; bidx0 = k; }
            // subtile 1
            dist = xn1 + wn - 2.0f * acc1[r];
            if (!(dist == dist)) dist = 0.f;
            dw = dist * rs * (1.0f / (float)DD);
            if (!(dw == dw)) dw = 0.f;
            act = rs / (rs + dw + 1e-7f);
            if (act > best1) { best1 = act; bidx1 = k; }
        }
    }

    // merge the two half-lanes that own the same column b (per subtile)
    float ov = __shfl_xor(best0, 16, 32);
    int   oi = __shfl_xor(bidx0, 16, 32);
    if (ov > best0 || (ov == best0 && oi < bidx0)) { best0 = ov; bidx0 = oi; }
    ov = __shfl_xor(best1, 16, 32);
    oi = __shfl_xor(bidx1, 16, 32);
    if (ov > best1 || (ov == best1 && oi < bidx1)) { best1 = ov; bidx1 = oi; }
    if (hi == 0) {
        pv[(size_t)chunk * BB + b0 + col]      = best0;
        pi[(size_t)chunk * BB + b0 + col]      = bidx0;
        pv[(size_t)chunk * BB + b0 + 16 + col] = best1;
        pi[(size_t)chunk * BB + b0 + 16 + col] = bidx1;
    }
}

// ---------------------------------------------------------------------------
// Kernel 3: reduce NCHUNK partial argmaxes per row (chunks ascend in k,
// so strict > keeps the first/lowest index on ties).
// ---------------------------------------------------------------------------
__global__ void som_argreduce(const float* __restrict__ pv,
                              const int* __restrict__ pi,
                              int* __restrict__ idx_ws,
                              int* __restrict__ idx_out)
{
    int b = blockIdx.x * blockDim.x + threadIdx.x;
    if (b >= BB) return;
    float best = pv[b];
    int   bi   = pi[b];
    #pragma unroll
    for (int c = 1; c < NCHUNK; ++c) {
        float v = pv[(size_t)c * BB + b];
        int   i = pi[(size_t)c * BB + b];
        if (v > best) { best = v; bi = i; }
    }
    idx_ws[b]  = bi;
    idx_out[b] = bi;
}

__global__ void som_winner_init(int* __restrict__ winner)
{
    int k = blockIdx.x * blockDim.x + threadIdx.x;
    if (k < KK) winner[k] = -1;
}

__global__ void som_winner_scatter(const int* __restrict__ idx,
                                   int* __restrict__ winner)
{
    int b = blockIdx.x * blockDim.x + threadIdx.x;
    if (b < BB) atomicMax(&winner[idx[b]], b);   // deterministic (max)
}

// ---------------------------------------------------------------------------
// Kernel 4: per-row delta sums for the loss (one wave32 per row).
// ---------------------------------------------------------------------------
__global__ __launch_bounds__(128) void som_rowdelta(
    const float* __restrict__ x, const float* __restrict__ w,
    const int* __restrict__ idx, float* __restrict__ rowsum)
{
    int lane = threadIdx.x & 31;
    int b    = (blockIdx.x * blockDim.x + threadIdx.x) >> 5;
    if (b >= BB) return;
    int k = idx[b];
    const float* xr = x + (size_t)b * DD;
    const float* wr = w + (size_t)k * DD;
    float s = 0.f;
    #pragma unroll 4
    for (int j = 0; j < DD / 32; ++j) {
        int d = lane + 32 * j;
        s += xr[d] - wr[d];
    }
    s = wave_sum(s);
    if (lane == 0) rowsum[b] = s;
}

// ---------------------------------------------------------------------------
// Kernel 5: scatter update for winning rows (one block per b).
// Reads ORIGINAL weights / moving_avg (d_in); writes into d_out regions that
// were pre-filled with pass-through copies.
// ---------------------------------------------------------------------------
__global__ __launch_bounds__(256) void som_update(
    const float* __restrict__ x, const float* __restrict__ w,
    const float* __restrict__ ma, const int* __restrict__ idx,
    const int* __restrict__ winner,
    float* __restrict__ out_w, float* __restrict__ out_ma,
    float* __restrict__ out_rel)
{
    int b = blockIdx.x;
    int k = idx[b];
    if (winner[k] != b) return;      // this b does not own the scatter slot

    const int t = threadIdx.x;
    float nv[4], xe[4], we[4];
    float mx = -3.4e38f, mn = 3.4e38f, sm = 0.f;
    #pragma unroll
    for (int j = 0; j < 4; ++j) {
        int d = t + j * 256;
        float xx = x[(size_t)b * DD + d];
        float ww = w[(size_t)k * DD + d];
        float nm = ACOEF * fabsf(xx - ww) + (1.0f - ACOEF) * ma[(size_t)k * DD + d];
        xe[j] = xx; we[j] = ww; nv[j] = nm;
        mx = fmaxf(mx, nm); mn = fminf(mn, nm); sm += nm;
    }
    __shared__ float smx[256], smn[256], ssm[256];
    smx[t] = mx; smn[t] = mn; ssm[t] = sm;
    __syncthreads();
    #pragma unroll
    for (int s = 128; s > 0; s >>= 1) {
        if (t < s) {
            smx[t] = fmaxf(smx[t], smx[t + s]);
            smn[t] = fminf(smn[t], smn[t + s]);
            ssm[t] += ssm[t + s];
        }
        __syncthreads();
    }
    mx = smx[0]; mn = smn[0];
    float avg = ssm[0] * (1.0f / (float)DD);
    float den = EPS_DS * (mx - mn);

    #pragma unroll
    for (int j = 0; j < 4; ++j) {
        int d = t + j * 256;
        float arg = (nv[j] - avg) / den;
        float r = 1.0f / (1.0f + expf(arg));
        if (!(r == r)) r = 1.0f;                 // NaN -> 1
        out_ma [(size_t)k * DD + d] = nv[j];
        out_rel[(size_t)k * DD + d] = r;
        out_w  [(size_t)k * DD + d] = we[j] + LR * (xe[j] - we[j]);
    }
}

// ---------------------------------------------------------------------------
// Kernel 6: deterministic fixed-order final loss reduction (single block).
// ---------------------------------------------------------------------------
__global__ __launch_bounds__(256) void som_loss(const float* __restrict__ rowsum,
                                                float* __restrict__ out)
{
    const int t = threadIdx.x;
    float s = 0.f;
    for (int i = t; i < BB; i += 256) s += rowsum[i];
    __shared__ float red[256];
    red[t] = s;
    __syncthreads();
    #pragma unroll
    for (int st = 128; st > 0; st >>= 1) {
        if (t < st) red[t] += red[t + st];
        __syncthreads();
    }
    if (t == 0) out[0] = red[0] * (LR / (float)BB);
}

// ---------------------------------------------------------------------------
extern "C" void kernel_launch(void* const* d_in, const int* in_sizes, int n_in,
                              void* d_out, int out_size, void* d_ws, size_t ws_size,
                              hipStream_t stream) {
    (void)in_sizes; (void)n_in; (void)out_size; (void)ws_size;
    const float* x   = (const float*)d_in[0];   // [B,D]
    const float* w   = (const float*)d_in[1];   // [K,D]
    const float* ma  = (const float*)d_in[2];   // [K,D]
    const float* rel = (const float*)d_in[3];   // [K,D]

    // ---- workspace layout ----
    char* ws = (char*)d_ws;
    unsigned short* xb = (unsigned short*)ws;                          // B*D bf16
    unsigned short* wb = (unsigned short*)(ws + (size_t)BB * DD * 2);  // K*D bf16
    float* x_norm  = (float*)(ws + (size_t)(BB + KK) * DD * 2);
    float* w_norm  = x_norm + BB;
    float* rel_sum = w_norm + KK;
    float* pv      = rel_sum + KK;                 // NCHUNK*B
    int*   pi      = (int*)(pv + (size_t)NCHUNK * BB);
    int*   idx_ws  = pi + (size_t)NCHUNK * BB;
    int*   winner  = idx_ws + BB;
    float* rowsum  = (float*)(winner + KK);

    // ---- output layout: loss | idx | weights | moving_avg | relevance ----
    float* out      = (float*)d_out;
    int*   out_idx  = (int*)(out + 1);
    float* out_w    = out + 1 + BB;
    float* out_ma   = out_w + (size_t)KK * DD;
    float* out_rel  = out_ma + (size_t)KK * DD;

    // 1) norms + bf16 conversion: (K+B) waves, 4 waves/block
    som_prep<<<(KK + BB) / 4, 128, 0, stream>>>(x, w, rel, xb, wb,
                                                x_norm, w_norm, rel_sum);

    // 2) WMMA BMU: 1024 waves (128 b-tiles x 8 k-chunks), 2 waves/block,
    //    128KB dynamic LDS (64KB x-tile per wave, TDM-loaded)
    som_bmu<<<(BB / BTILE) * NCHUNK / BMU_WPB, 64,
              BMU_WPB * BTILE * DD * 2, stream>>>(
        xb, wb, x_norm, w_norm, rel_sum, pv, pi);

    // 3) argmax reduce across k-chunks
    som_argreduce<<<BB / 256, 256, 0, stream>>>(pv, pi, idx_ws, out_idx);

    // 4) winner-per-slot (highest b wins, deterministic)
    som_winner_init<<<KK / 256, 256, 0, stream>>>(winner);
    som_winner_scatter<<<BB / 256, 256, 0, stream>>>(idx_ws, winner);

    // 5) pass-through copies of the three state arrays
    const size_t bytesKD = (size_t)KK * DD * sizeof(float);
    (void)hipMemcpyAsync(out_w,   w,   bytesKD, hipMemcpyDeviceToDevice, stream);
    (void)hipMemcpyAsync(out_ma,  ma,  bytesKD, hipMemcpyDeviceToDevice, stream);
    (void)hipMemcpyAsync(out_rel, rel, bytesKD, hipMemcpyDeviceToDevice, stream);

    // 6) loss row sums (uses ORIGINAL weights)
    som_rowdelta<<<BB / 4, 128, 0, stream>>>(x, w, idx_ws, rowsum);

    // 7) scatter updates for winning rows
    som_update<<<BB, 256, 0, stream>>>(x, w, ma, idx_ws, winner,
                                       out_w, out_ma, out_rel);

    // 8) final deterministic loss
    som_loss<<<1, 256, 0, stream>>>(rowsum, out);
}
